// FNetBlock_87445534146971
// MI455X (gfx1250) — matrix operand order repaired and checked by reading, not compiled, not run on hardware
//
#include <hip/hip_runtime.h>
#include <cstdint>

// ---------------------------------------------------------------------------
// FNet block for MI455X (gfx1250, wave32, WMMA).
// attn = Re(FFT2(x)) computed as DFT GEMMs:  C_S @ (X @ C_D) - S_S @ (X @ S_D)
// GEMMs run on v_wmma_f32_16x16x32_bf16, double-buffered LDS, A tiles staged
// with async global->LDS (ASYNCcnt), B tiles via b128 loads + packed b32
// transpose stores into LDS.
// ---------------------------------------------------------------------------

#define USE_ASYNC 1

typedef __attribute__((ext_vector_type(16))) __bf16 v16bf;
typedef __attribute__((ext_vector_type(8)))  float  v8f;

__device__ __forceinline__ unsigned short f2bf(float x) {
  union { float f; uint32_t u; } v; v.f = x;
  uint32_t r = v.u + 0x7FFFu + ((v.u >> 16) & 1u);   // round-to-nearest-even
  return (unsigned short)(r >> 16);
}

__device__ __forceinline__ float gelu_exact(float x) {
  return 0.5f * x * (1.0f + erff(x * 0.70710678118654752440f));
}

__device__ __forceinline__ void async_wait_all() {
#if USE_ASYNC
  asm volatile("s_wait_asynccnt 0x0" ::: "memory");
#endif
}

// ----------------------------- epilogue modes ------------------------------
enum {
  MODE_STORE_BF16     = 0,  // outBf = acc
  MODE_STORE_F32      = 1,  // outF  = alpha*acc
  MODE_ACCUM_F32      = 2,  // outF += alpha*acc
  MODE_BIAS_GELU_BF16 = 3,  // outBf = gelu(acc + bias[n])
  MODE_BIAS_RES_F32   = 4,  // outF  = acc + bias[n] + res[m*N+n]
};

// ------------------------- generic bf16 WMMA GEMM --------------------------
// C(MxN) = A(MxK, row-major bf16) * B(KxN, row-major bf16), batched via z.
// Block tile 128x128, 8 waves (4 M x 2 N), wave tile 32x64 = 2x4 WMMA accums.
#define BM 128
#define BN 128
#define BK 32
#define LDSP 40   // padded K pitch (elements); row pitch 80B = multiple of 16B

template <int MODE>
__global__ __launch_bounds__(256) void gemm_bf16_wmma(
    const unsigned short* __restrict__ A, long long lda, long long strideA,
    const unsigned short* __restrict__ Bm, long long ldb, long long strideB,
    float* __restrict__ Cf, unsigned short* __restrict__ Cbf, long long strideC,
    const float* __restrict__ bias, const float* __restrict__ res,
    float alpha, int M, int N, int K) {
  __shared__ unsigned short As[2][BM * LDSP];
  __shared__ unsigned short Bs[2][BN * LDSP];

  const int tid  = threadIdx.x;
  const int lane = tid & 31;
  const int wave = tid >> 5;
  const int wm   = wave & 3;   // 0..3 : 32-row strip
  const int wn   = wave >> 2;  // 0..1 : 64-col strip
  const int l15  = lane & 15;
  const int half = lane >> 4;

  const int m0 = blockIdx.y * BM;
  const int n0 = blockIdx.x * BN;
  const int zb = blockIdx.z;

  const unsigned short* Ab = A  + (long long)zb * strideA + (long long)m0 * lda;
  const unsigned short* Bb = Bm + (long long)zb * strideB;
  const long long coff = (long long)zb * strideC;

  const v8f vzero = {0.f,0.f,0.f,0.f,0.f,0.f,0.f,0.f};
  v8f acc[2][4];
#pragma unroll
  for (int i = 0; i < 2; ++i)
#pragma unroll
    for (int j = 0; j < 4; ++j) acc[i][j] = vzero;

  union Frag { v16bf v; unsigned int u[8]; };
  union BV   { uint4 v; unsigned short s[8]; };

  // B staging geometry: each thread owns a 2(k) x 8(n) sub-block of the
  // 32x128 tile: rows bk,bk+1 and columns bc..bc+7.
  const int bk = (tid >> 4) * 2;   // 0,2,...,30
  const int bc = (tid & 15) * 8;   // 0,8,...,120

  // ---- A tile stage: 128x32, one b128 per thread x2, async direct to LDS
  auto stage_a = [&](int buf, int kb) {
#pragma unroll
    for (int it = 0; it < 2; ++it) {
      int idx = (tid + it * 256) * 8;       // over 128*32 elements
      int r = idx >> 5, c = idx & 31;
      const unsigned short* g = Ab + (long long)r * lda + kb + c;
      unsigned short* l = &As[buf][r * LDSP + c];
#if USE_ASYNC
      unsigned long long ga = (unsigned long long)g;
      unsigned la = (unsigned)(size_t)l;
      asm volatile("global_load_async_to_lds_b128 %0, %1, off"
                   :: "v"(la), "v"(ga) : "memory");
#else
      *(uint4*)l = *(const uint4*)g;
#endif
    }
  };
  // ---- B tile: two coalesced b128 row loads (rows bk and bk+1)
  auto load_b = [&](int kb, BV& lo, BV& hi) {
    lo.v = *(const uint4*)(Bb + (long long)(kb + bk)     * ldb + n0 + bc);
    hi.v = *(const uint4*)(Bb + (long long)(kb + bk + 1) * ldb + n0 + bc);
  };
  // ---- transpose-store into Bs[n][k]: pack (k,k+1) per column -> b32 store
  auto store_b = [&](int buf, const BV& lo, const BV& hi) {
#pragma unroll
    for (int j = 0; j < 8; ++j) {
      unsigned int wv = (unsigned int)lo.s[j] | ((unsigned int)hi.s[j] << 16);
      *(unsigned int*)&Bs[buf][(bc + j) * LDSP + bk] = wv;
    }
  };

  // ---- prologue: stage K-tile 0 into buffer 0
  {
    stage_a(0, 0);
    BV lo, hi;
    load_b(0, lo, hi);
    store_b(0, lo, hi);
    async_wait_all();
  }
  __syncthreads();

  int cur = 0;
  for (int kb = 0; kb < K; kb += BK) {
    const int nxt = kb + BK;
    BV blo, bhi;
    if (nxt < K) {
      stage_a(cur ^ 1, nxt);                 // async -> other buffer
      load_b(nxt, blo, bhi);                 // global -> regs
      // prefetch the tile after next into L2 (global_prefetch_b8)
      if (nxt + BK < K) {
        __builtin_prefetch(Ab + (long long)(tid >> 1) * lda + nxt + BK, 0, 0);
        __builtin_prefetch(Bb + (long long)(nxt + BK + bk) * ldb + n0 + bc, 0, 0);
      }
    }

    // ---- gather fragments (CDNA5 16-bit WMMA operand layouts)
    Frag a[2], b[4];
#pragma unroll
    for (int p = 0; p < 8; ++p) {
      int ka  = ((p < 4) ? (2 * p) : (2 * p + 8)) + 8 * half;  // A elem-pair K
      int kbb = ka;                                            // B elem-pair K
#pragma unroll
      for (int sm = 0; sm < 2; ++sm)
        a[sm].u[p] = *(const unsigned int*)&As[cur][(wm * 32 + sm * 16 + l15) * LDSP + ka];
#pragma unroll
      for (int sn = 0; sn < 4; ++sn)
        b[sn].u[p] = *(const unsigned int*)&Bs[cur][(wn * 64 + sn * 16 + l15) * LDSP + kbb];
    }
#pragma unroll
    for (int sm = 0; sm < 2; ++sm)
#pragma unroll
      for (int sn = 0; sn < 4; ++sn)
        acc[sm][sn] = __builtin_amdgcn_wmma_f32_16x16x32_bf16(
            false, a[sm].v, false, b[sn].v, (short)0, acc[sm][sn], false, false);

    if (nxt < K) store_b(cur ^ 1, blo, bhi);
    async_wait_all();
    __syncthreads();
    cur ^= 1;
  }

  // ---- epilogue: C/D layout m = base + r + 8*half, n = base + l15
#pragma unroll
  for (int sm = 0; sm < 2; ++sm)
#pragma unroll
    for (int sn = 0; sn < 4; ++sn)
#pragma unroll
      for (int r = 0; r < 8; ++r) {
        long long m = (long long)m0 + wm * 32 + sm * 16 + r + 8 * half;
        long long n = (long long)n0 + wn * 64 + sn * 16 + l15;
        long long idx = coff + m * (long long)N + n;
        float v = acc[sm][sn][r] * alpha;
        if (MODE == MODE_STORE_BF16) {
          Cbf[idx] = f2bf(v);
        } else if (MODE == MODE_STORE_F32) {
          Cf[idx] = v;
        } else if (MODE == MODE_ACCUM_F32) {
          Cf[idx] += v;
        } else if (MODE == MODE_BIAS_GELU_BF16) {
          Cbf[idx] = f2bf(gelu_exact(v + bias[n]));
        } else if (MODE == MODE_BIAS_RES_F32) {
          Cf[idx] = v + bias[n] + res[m * (long long)N + n];
        }
      }
}

// ----------------------------- prep kernels --------------------------------
__global__ void conv_f32_bf16(const float* __restrict__ src,
                              unsigned short* __restrict__ dst, long long n) {
  long long i = (long long)blockIdx.x * blockDim.x + threadIdx.x;
  long long stride = (long long)gridDim.x * blockDim.x;
  for (; i < n; i += stride) dst[i] = f2bf(src[i]);
}

// DFT matrices: C[j,k]=cos(2*pi*j*k/N), S[j,k]=sin(2*pi*j*k/N)
__global__ void build_dft(unsigned short* __restrict__ C,
                          unsigned short* __restrict__ S, int N) {
  int k = blockIdx.x * 16 + threadIdx.x;
  int j = blockIdx.y * 16 + threadIdx.y;
  if (j >= N || k >= N) return;
  long long t = ((long long)j * (long long)k) % N;
  float ang = 6.28318530717958647692f * ((float)t / (float)N);
  float s, c;
  __sincosf(ang, &s, &c);
  C[(long long)j * N + k] = f2bf(c);
  S[(long long)j * N + k] = f2bf(s);
}

// ------------------------------- layernorm ---------------------------------
__global__ __launch_bounds__(256) void layernorm_row(
    const float* __restrict__ A, const float* __restrict__ Badd,
    const float* __restrict__ gamma, const float* __restrict__ beta,
    float* __restrict__ outF, unsigned short* __restrict__ outBf, int D) {
  __shared__ float s1[256], s2[256];
  const long long row = blockIdx.x;
  const int t = threadIdx.x;
  const int nit = (D + 255) / 256;
  float vloc[8];
  float p1 = 0.f, p2 = 0.f;
  for (int i = 0; i < nit; ++i) {
    int c = t + i * 256;
    float v = 0.f;
    if (c < D) {
      v = A[row * D + c];
      if (Badd) v += Badd[row * D + c];
    }
    vloc[i] = v; p1 += v; p2 += v * v;
  }
  s1[t] = p1; s2[t] = p2;
  __syncthreads();
  for (int off = 128; off > 0; off >>= 1) {
    if (t < off) { s1[t] += s1[t + off]; s2[t] += s2[t + off]; }
    __syncthreads();
  }
  float mu  = s1[0] / (float)D;
  float var = s2[0] / (float)D - mu * mu;
  float rstd = rsqrtf(var + 1e-6f);
  for (int i = 0; i < nit; ++i) {
    int c = t + i * 256;
    if (c < D) {
      float y = (vloc[i] - mu) * rstd * gamma[c] + beta[c];
      outF[row * D + c] = y;
      if (outBf) outBf[row * D + c] = f2bf(y);
    }
  }
}

// ------------------------------- launcher ----------------------------------
extern "C" void kernel_launch(void* const* d_in, const int* in_sizes, int n_in,
                              void* d_out, int out_size, void* d_ws, size_t ws_size,
                              hipStream_t stream) {
  (void)in_sizes; (void)n_in; (void)out_size; (void)ws_size;
  constexpr long long Bz = 8, Sq = 4096, Dm = 768, Fm = 3072;
  constexpr long long BS = Bz * Sq;

  const float* x   = (const float*)d_in[0];
  const float* W1  = (const float*)d_in[1];
  const float* b1  = (const float*)d_in[2];
  const float* W2  = (const float*)d_in[3];
  const float* b2  = (const float*)d_in[4];
  const float* g1  = (const float*)d_in[5];
  const float* be1 = (const float*)d_in[6];
  const float* g2  = (const float*)d_in[7];
  const float* be2 = (const float*)d_in[8];
  float* out = (float*)d_out;

  // workspace carve-out (256B aligned)
  uint8_t* w = (uint8_t*)d_ws;
  size_t cur = 0;
  auto alloc = [&](size_t bytes) -> void* {
    void* p = w + cur;
    cur += (bytes + 255) & ~(size_t)255;
    return p;
  };
  unsigned short* Xbf   = (unsigned short*)alloc(2 * BS * Dm);
  unsigned short* W1bf  = (unsigned short*)alloc(2 * Dm * Fm);
  unsigned short* W2bf  = (unsigned short*)alloc(2 * Fm * Dm);
  unsigned short* Cd    = (unsigned short*)alloc(2 * Dm * Dm);
  unsigned short* Sd    = (unsigned short*)alloc(2 * Dm * Dm);
  unsigned short* Cs    = (unsigned short*)alloc(2 * Sq * Sq);
  unsigned short* Ss    = (unsigned short*)alloc(2 * Sq * Sq);
  unsigned short* P     = (unsigned short*)alloc(2 * BS * Dm);
  unsigned short* Q     = (unsigned short*)alloc(2 * BS * Dm);
  float*          attn  = (float*)alloc(4 * BS * Dm);   // reused as ffn_pre
  float*          out1f = (float*)alloc(4 * BS * Dm);
  unsigned short* out1b = (unsigned short*)alloc(2 * BS * Dm);
  unsigned short* h     = (unsigned short*)alloc(2 * BS * Fm);

  // 1) bf16 conversions
  conv_f32_bf16<<<4096, 256, 0, stream>>>(x,  Xbf,  BS * Dm);
  conv_f32_bf16<<<2048, 256, 0, stream>>>(W1, W1bf, Dm * Fm);
  conv_f32_bf16<<<2048, 256, 0, stream>>>(W2, W2bf, Fm * Dm);

  // 2) DFT twiddle matrices
  {
    dim3 blk(16, 16);
    build_dft<<<dim3(Dm / 16, Dm / 16), blk, 0, stream>>>(Cd, Sd, (int)Dm);
    build_dft<<<dim3(Sq / 16, Sq / 16), blk, 0, stream>>>(Cs, Ss, (int)Sq);
  }

  // 3) P = X @ C_D ; Q = X @ S_D   (per batch, M=S, N=D, K=D)
  {
    dim3 grid(Dm / BN, Sq / BM, Bz);
    gemm_bf16_wmma<MODE_STORE_BF16><<<grid, 256, 0, stream>>>(
        Xbf, Dm, Sq * Dm, Cd, Dm, 0, nullptr, P, Sq * Dm,
        nullptr, nullptr, 1.0f, (int)Sq, (int)Dm, (int)Dm);
    gemm_bf16_wmma<MODE_STORE_BF16><<<grid, 256, 0, stream>>>(
        Xbf, Dm, Sq * Dm, Sd, Dm, 0, nullptr, Q, Sq * Dm,
        nullptr, nullptr, 1.0f, (int)Sq, (int)Dm, (int)Dm);
  }

  // 4) attn = C_S @ P ; attn -= S_S @ Q   (per batch, M=S, N=D, K=S)
  {
    dim3 grid(Dm / BN, Sq / BM, Bz);
    gemm_bf16_wmma<MODE_STORE_F32><<<grid, 256, 0, stream>>>(
        Cs, Sq, 0, P, Dm, Sq * Dm, attn, nullptr, Sq * Dm,
        nullptr, nullptr, 1.0f, (int)Sq, (int)Dm, (int)Sq);
    gemm_bf16_wmma<MODE_ACCUM_F32><<<grid, 256, 0, stream>>>(
        Ss, Sq, 0, Q, Dm, Sq * Dm, attn, nullptr, Sq * Dm,
        nullptr, nullptr, -1.0f, (int)Sq, (int)Dm, (int)Sq);
  }

  // 5) out1 = LN(x + attn)  -> f32 + bf16
  layernorm_row<<<(unsigned)BS, 256, 0, stream>>>(x, attn, g1, be1, out1f, out1b, (int)Dm);

  // 6) h = gelu(out1 @ W1 + b1)   (M=BS, N=F, K=D)
  {
    dim3 grid(Fm / BN, BS / BM, 1);
    gemm_bf16_wmma<MODE_BIAS_GELU_BF16><<<grid, 256, 0, stream>>>(
        out1b, Dm, 0, W1bf, Fm, 0, nullptr, h, 0,
        b1, nullptr, 1.0f, (int)BS, (int)Fm, (int)Dm);
  }

  // 7) ffn_pre = out1 + (h @ W2 + b2)   (M=BS, N=D, K=F), reuse attn buffer
  {
    dim3 grid(Dm / BN, BS / BM, 1);
    gemm_bf16_wmma<MODE_BIAS_RES_F32><<<grid, 256, 0, stream>>>(
        h, Fm, 0, W2bf, Dm, 0, attn, nullptr, 0,
        b2, out1f, 1.0f, (int)BS, (int)Dm, (int)Fm);
  }

  // 8) out = LN(ffn_pre)
  layernorm_row<<<(unsigned)BS, 256, 0, stream>>>(attn, nullptr, g2, be2, out, nullptr, (int)Dm);
}
